// EnhancedClassifier_2946347565680
// MI455X (gfx1250) — matrix operand order repaired and checked by reading, not compiled, not run on hardware
//
#include <hip/hip_runtime.h>

#define D_IN  512
#define D_HID 128

typedef __bf16 bf16;
typedef __attribute__((ext_vector_type(16))) __bf16 v16bf;
typedef __attribute__((ext_vector_type(8)))  float  v8f;
typedef __attribute__((ext_vector_type(4)))  unsigned int u32x4;

union FragAB { v16bf v; u32x4 q[2]; };

// ---------------- small utility kernels ----------------

__global__ void k_fill1(float* __restrict__ p, int n) {
    int i = blockIdx.x * blockDim.x + threadIdx.x;
    if (i < n) p[i] = 1.0f;
}

__global__ void k_deg_accum(const int* __restrict__ dst, float* __restrict__ deg, int E) {
    int e = blockIdx.x * blockDim.x + threadIdx.x;
    if (e < E) atomicAdd(&deg[dst[e]], 1.0f);
}

__global__ void k_rsqrt_inplace(float* __restrict__ p, int n) {
    int i = blockIdx.x * blockDim.x + threadIdx.x;
    if (i < n) p[i] = rsqrtf(p[i]);
}

__global__ void k_convert_bf16(const float* __restrict__ in, bf16* __restrict__ out, int n) {
    int i = blockIdx.x * blockDim.x + threadIdx.x;
    if (i < n) out[i] = (bf16)in[i];
}

// W: [K,N] row-major f32  ->  Wt: [N,K] row-major bf16
__global__ void k_transpose_bf16(const float* __restrict__ W, bf16* __restrict__ Wt, int K, int N) {
    int t = blockIdx.x * blockDim.x + threadIdx.x;
    if (t >= K * N) return;
    int n = t / K, k = t - n * K;
    Wt[t] = (bf16)W[k * N + n];
}

// ---------------- WMMA GEMM: C[M,128] = A[M,K](bf16) * Bt[128,K]^T(bf16) ----------------
// block = 256 threads = 8 waves; wave w -> N-tile w (16 cols); blockIdx.x -> M-tile (16 rows)
__global__ void __launch_bounds__(256) k_gemm_bf16_wmma(
    const bf16* __restrict__ A, const bf16* __restrict__ Bt,
    float* __restrict__ C, int K, int nRows)
{
    const int lane = threadIdx.x & 31;
    const int wave = threadIdx.x >> 5;      // 0..7 -> n tile
    const int half = lane >> 4;             // 0 / 1
    const int mr   = lane & 15;
    const int m0   = blockIdx.x * 16;
    const int n0   = wave * 16;

    int arowi = m0 + mr; if (arowi >= nRows) arowi = nRows - 1;   // clamp (no divergence)
    const bf16* arow = A  + (size_t)arowi      * K;
    const bf16* brow = Bt + (size_t)(n0 + mr)  * K;

    v8f acc = {};
    for (int k0 = 0; k0 < K; k0 += 32) {
        FragAB a, b;
        const u32x4* ab = (const u32x4*)(arow + k0);
        const u32x4* bb = (const u32x4*)(brow + k0);
        // A 16x32 bf16 layout: lanes0-15 K={0..7,16..23}, lanes16-31 K={8..15,24..31}
        a.q[0] = ab[half];
        a.q[1] = ab[2 + half];
        // B 32x16 bf16 layout: lanes0-15 K={0..7,16..23}, lanes16-31 K={8..15,24..31} (col = lane&15)
        b.q[0] = bb[half];
        b.q[1] = bb[2 + half];
        acc = __builtin_amdgcn_wmma_f32_16x16x32_bf16(
            false, a.v, false, b.v, (short)0, acc, false, false);
    }

    // C/D layout: VGPR r -> row m0 + r + half*8, col n0 + (lane&15)
    float* crow = C + (size_t)(m0 + half * 8) * D_HID + n0 + mr;
    #pragma unroll
    for (int r = 0; r < 8; ++r) {
        int row = m0 + half * 8 + r;
        if (row < nRows) crow[(size_t)r * D_HID] = acc[r];
    }
}

// ---------------- GCN aggregation ----------------

// agg[n][f] = xw[n][f]*dinv[n]^2 + bias[f]
__global__ void k_self_init(const float* __restrict__ xw, const float* __restrict__ dinv,
                            const float* __restrict__ bias, float* __restrict__ agg, int total) {
    int t = blockIdx.x * blockDim.x + threadIdx.x;
    if (t >= total) return;
    int n = t >> 7, f = t & 127;
    float di = dinv[n];
    agg[t] = xw[t] * di * di + bias[f];
}

// one wave per edge, each lane handles 4 features (float4 gather + 4 f32 atomics)
__global__ void __launch_bounds__(256) k_scatter(const int* __restrict__ src, const int* __restrict__ dst,
                          const float* __restrict__ dinv, const float* __restrict__ xw,
                          float* __restrict__ agg, int E) {
    int t = blockIdx.x * blockDim.x + threadIdx.x;
    int e = t >> 5;
    if (e >= E) return;
    int lane = t & 31;
    int s = src[e], d = dst[e];
    float w = dinv[s] * dinv[d];
    const float4 xv = *(const float4*)(xw + (size_t)s * D_HID + lane * 4);
    float* o = agg + (size_t)d * D_HID + lane * 4;
    atomicAdd(o + 0, xv.x * w);
    atomicAdd(o + 1, xv.y * w);
    atomicAdd(o + 2, xv.z * w);
    atomicAdd(o + 3, xv.w * w);
}

// wave-per-node LayerNorm + ReLU, emit bf16 for next GEMM
__global__ void __launch_bounds__(256) k_ln_relu_bf16(const float* __restrict__ agg,
                             const float* __restrict__ gamma, const float* __restrict__ beta,
                             bf16* __restrict__ out, int nNodes) {
    int wave = threadIdx.x >> 5, lane = threadIdx.x & 31;
    int node = blockIdx.x * 8 + wave;
    if (node >= nNodes) return;
    const float4 v = *(const float4*)(agg + (size_t)node * D_HID + lane * 4);
    float s  = v.x + v.y + v.z + v.w;
    float ss = v.x * v.x + v.y * v.y + v.z * v.z + v.w * v.w;
    #pragma unroll
    for (int off = 16; off; off >>= 1) {
        s  += __shfl_xor(s,  off);
        ss += __shfl_xor(ss, off);
    }
    float mean = s * (1.0f / 128.0f);
    float var  = ss * (1.0f / 128.0f) - mean * mean;
    float rstd = rsqrtf(var + 1e-5f);
    int f = lane * 4;
    union { bf16 h[4]; unsigned long long u; } o;
    o.h[0] = (bf16)fmaxf((v.x - mean) * rstd * gamma[f + 0] + beta[f + 0], 0.0f);
    o.h[1] = (bf16)fmaxf((v.y - mean) * rstd * gamma[f + 1] + beta[f + 1], 0.0f);
    o.h[2] = (bf16)fmaxf((v.z - mean) * rstd * gamma[f + 2] + beta[f + 2], 0.0f);
    o.h[3] = (bf16)fmaxf((v.w - mean) * rstd * gamma[f + 3] + beta[f + 3], 0.0f);
    *(unsigned long long*)(out + (size_t)node * D_HID + f) = o.u;
}

__global__ void k_relu_bf16(const float* __restrict__ agg, bf16* __restrict__ out, int total) {
    int i = blockIdx.x * blockDim.x + threadIdx.x;
    if (i < total) out[i] = (bf16)fmaxf(agg[i], 0.0f);
}

// ---------------- layer 4 (128 -> 1) ----------------

__global__ void __launch_bounds__(256) k_dot_w4(const bf16* __restrict__ h, const float* __restrict__ W4,
                         float* __restrict__ xw4, int nNodes) {
    int wave = threadIdx.x >> 5, lane = threadIdx.x & 31;
    int node = blockIdx.x * 8 + wave;
    if (node >= nNodes) return;
    const bf16* row = h + (size_t)node * D_HID + lane * 4;
    float s = 0.0f;
    #pragma unroll
    for (int j = 0; j < 4; ++j) s += (float)row[j] * W4[lane * 4 + j];
    #pragma unroll
    for (int off = 16; off; off >>= 1) s += __shfl_xor(s, off);
    if (lane == 0) xw4[node] = s;
}

__global__ void k_out_self(const float* __restrict__ xw4, const float* __restrict__ dinv,
                           const float* __restrict__ b4, float* __restrict__ out, int n) {
    int i = blockIdx.x * blockDim.x + threadIdx.x;
    if (i < n) out[i] = xw4[i] * dinv[i] * dinv[i] + b4[0];
}

__global__ void k_scatter1(const int* __restrict__ src, const int* __restrict__ dst,
                           const float* __restrict__ dinv, const float* __restrict__ xw4,
                           float* __restrict__ out, int E) {
    int e = blockIdx.x * blockDim.x + threadIdx.x;
    if (e < E) {
        int s = src[e], d = dst[e];
        atomicAdd(&out[d], xw4[s] * dinv[s] * dinv[d]);
    }
}

// ---------------- host launcher ----------------

extern "C" void kernel_launch(void* const* d_in, const int* in_sizes, int n_in,
                              void* d_out, int out_size, void* d_ws, size_t ws_size,
                              hipStream_t stream) {
    const float* x      = (const float*)d_in[0];
    const float* W1     = (const float*)d_in[1];
    const float* b1     = (const float*)d_in[2];
    const float* gamma1 = (const float*)d_in[3];
    const float* beta1  = (const float*)d_in[4];
    const float* W2     = (const float*)d_in[5];
    const float* b2     = (const float*)d_in[6];
    const float* W3     = (const float*)d_in[7];
    const float* b3     = (const float*)d_in[8];
    const float* W4     = (const float*)d_in[9];
    const float* b4     = (const float*)d_in[10];
    const int*   ei     = (const int*)d_in[11];

    const int N = in_sizes[0] / D_IN;     // 50000
    const int E = in_sizes[11] / 2;       // 625000
    const int* src = ei;
    const int* dst = ei + E;

    char* ws = (char*)d_ws;
    size_t off = 0;
    auto take = [&](size_t bytes) -> char* {
        char* p = ws + off;
        off = (off + bytes + 255) & ~(size_t)255;
        return p;
    };

    bf16*  bfA  = (bf16*)take((size_t)N * D_IN  * sizeof(bf16));   // activations (bf16), reused
    bf16*  bfW  = (bf16*)take((size_t)D_IN * D_HID * sizeof(bf16)); // transposed weights (bf16), reused
    float* xw   = (float*)take((size_t)N * D_HID * sizeof(float));
    float* agg  = (float*)take((size_t)N * D_HID * sizeof(float));
    float* dinv = (float*)take((size_t)N * sizeof(float));
    float* out  = (float*)d_out;

    const int T = 256;
    const unsigned gN    = (N + T - 1) / T;
    const unsigned gE    = (E + T - 1) / T;
    const unsigned gNF   = ((unsigned)((size_t)N * D_HID + T - 1)) / T;
    const unsigned gEdge = (unsigned)(((size_t)E * 32 + T - 1) / T);
    const unsigned gWave = (N + 7) / 8;
    const unsigned gTile = (N + 15) / 16;

    // degrees (with self-loop) -> dinv
    k_fill1<<<gN, T, 0, stream>>>(dinv, N);
    k_deg_accum<<<gE, T, 0, stream>>>(dst, dinv, E);
    k_rsqrt_inplace<<<gN, T, 0, stream>>>(dinv, N);

    // ---- layer 1: 512 -> 128, LN + ReLU
    k_convert_bf16<<<(unsigned)(((size_t)N * D_IN + T - 1) / T), T, 0, stream>>>(x, bfA, N * D_IN);
    k_transpose_bf16<<<(D_IN * D_HID + T - 1) / T, T, 0, stream>>>(W1, bfW, D_IN, D_HID);
    k_gemm_bf16_wmma<<<gTile, T, 0, stream>>>(bfA, bfW, xw, D_IN, N);
    k_self_init<<<gNF, T, 0, stream>>>(xw, dinv, b1, agg, N * D_HID);
    k_scatter<<<gEdge, T, 0, stream>>>(src, dst, dinv, xw, agg, E);
    k_ln_relu_bf16<<<gWave, T, 0, stream>>>(agg, gamma1, beta1, bfA, N);

    // ---- layer 2: 128 -> 128, ReLU
    k_transpose_bf16<<<(D_HID * D_HID + T - 1) / T, T, 0, stream>>>(W2, bfW, D_HID, D_HID);
    k_gemm_bf16_wmma<<<gTile, T, 0, stream>>>(bfA, bfW, xw, D_HID, N);
    k_self_init<<<gNF, T, 0, stream>>>(xw, dinv, b2, agg, N * D_HID);
    k_scatter<<<gEdge, T, 0, stream>>>(src, dst, dinv, xw, agg, E);
    k_relu_bf16<<<gNF, T, 0, stream>>>(agg, bfA, N * D_HID);

    // ---- layer 3: 128 -> 128, ReLU
    k_transpose_bf16<<<(D_HID * D_HID + T - 1) / T, T, 0, stream>>>(W3, bfW, D_HID, D_HID);
    k_gemm_bf16_wmma<<<gTile, T, 0, stream>>>(bfA, bfW, xw, D_HID, N);
    k_self_init<<<gNF, T, 0, stream>>>(xw, dinv, b3, agg, N * D_HID);
    k_scatter<<<gEdge, T, 0, stream>>>(src, dst, dinv, xw, agg, E);
    k_relu_bf16<<<gNF, T, 0, stream>>>(agg, bfA, N * D_HID);

    // ---- layer 4: 128 -> 1 (dot + aggregate directly into d_out)
    float* xw4 = xw;
    k_dot_w4<<<gWave, T, 0, stream>>>(bfA, W4, xw4, N);
    k_out_self<<<gN, T, 0, stream>>>(xw4, dinv, b4, out, N);
    k_scatter1<<<gE, T, 0, stream>>>(src, dst, dinv, xw4, out, E);
}
